// HierarchicalFeatureNet_44994077393250
// MI455X (gfx1250) — matrix operand order repaired and checked by reading, not compiled, not run on hardware
//
#include <hip/hip_runtime.h>
#include <hip/hip_bf16.h>

typedef __attribute__((ext_vector_type(16))) _Float16 v16h;
typedef __attribute__((ext_vector_type(8)))  _Float16 v8h;
typedef __attribute__((ext_vector_type(8)))  float    v8f;

#define NRES 8192
#define MNOD 4096   // NRES * 0.5
#define CCH  256
#define CUT3 0.027f // 0.3^3
#define NEGBIG (-1.0e30f)

// ---------------------------------------------------------------------------
// 1) Farthest-point sampling: single workgroup, dist[] in LDS, block argmax.
//    Produces node positions in ascending-index order (matches nonzero()).
// ---------------------------------------------------------------------------
__global__ __launch_bounds__(1024)
void hfn_fps_kernel(const float* __restrict__ pos,
                    float* __restrict__ nodepos_ws,  // (M,3) workspace
                    float* __restrict__ nodepos_out) // (M,3) in d_out
{
    __shared__ float sdist[NRES];            // 32 KB
    __shared__ unsigned char ssel[NRES];     // 8 KB
    __shared__ float s_val[1024];
    __shared__ int   s_idx[1024];
    __shared__ int   s_cnt[1024];

    const int tid = threadIdx.x;
    const float p0x = pos[0], p0y = pos[1], p0z = pos[2];

    for (int i = tid; i < NRES; i += 1024) {
        float dx = pos[3*i+0] - p0x;
        float dy = pos[3*i+1] - p0y;
        float dz = pos[3*i+2] - p0z;
        float d  = sqrtf(dx*dx + dy*dy + dz*dz);
        sdist[i] = (i == 0) ? NEGBIG : d;
        ssel[i]  = (i == 0) ? 1 : 0;
    }
    __syncthreads();

    for (int it = 1; it < MNOD; ++it) {
        // block-wide argmax with first-index tie-break (contiguous chunks)
        float best = NEGBIG; int bidx = 0x7fffffff;
        const int base = tid * (NRES / 1024);
        for (int i = base; i < base + (NRES / 1024); ++i) {
            float v = sdist[i];
            if (v > best) { best = v; bidx = i; }
        }
        s_val[tid] = best; s_idx[tid] = bidx;
        __syncthreads();
        for (int s = 512; s > 0; s >>= 1) {
            if (tid < s) {
                float ov = s_val[tid + s]; int oi = s_idx[tid + s];
                if (ov > s_val[tid] || (ov == s_val[tid] && oi < s_idx[tid])) {
                    s_val[tid] = ov; s_idx[tid] = oi;
                }
            }
            __syncthreads();
        }
        const int newn = s_idx[0];
        const float nx = pos[3*newn+0], ny = pos[3*newn+1], nz = pos[3*newn+2];
        if (tid == 0) ssel[newn] = 1;
        for (int i = base; i < base + (NRES / 1024); ++i) {
            float cur = sdist[i];
            if (cur != NEGBIG) {
                float dx = pos[3*i+0] - nx;
                float dy = pos[3*i+1] - ny;
                float dz = pos[3*i+2] - nz;
                float d  = sqrtf(dx*dx + dy*dy + dz*dz);
                float nd = fminf(cur, d);
                sdist[i] = (i == newn) ? NEGBIG : nd;
            }
        }
        __syncthreads();
    }

    // compact selected indices (ascending) -> node positions
    const int chunk = NRES / 1024;
    int cnt = 0;
    for (int i = tid * chunk; i < tid * chunk + chunk; ++i) cnt += ssel[i];
    s_cnt[tid] = cnt;
    __syncthreads();
    // inclusive Hillis-Steele scan
    for (int off = 1; off < 1024; off <<= 1) {
        int v = (tid >= off) ? s_cnt[tid - off] : 0;
        __syncthreads();
        s_cnt[tid] += v;
        __syncthreads();
    }
    int rank = s_cnt[tid] - cnt; // exclusive
    for (int i = tid * chunk; i < tid * chunk + chunk; ++i) {
        if (ssel[i]) {
            nodepos_ws[3*rank+0]  = pos[3*i+0];
            nodepos_ws[3*rank+1]  = pos[3*i+1];
            nodepos_ws[3*rank+2]  = pos[3*i+2];
            nodepos_out[3*rank+0] = pos[3*i+0];
            nodepos_out[3*rank+1] = pos[3*i+1];
            nodepos_out[3*rank+2] = pos[3*i+2];
            rank++;
        }
    }
}

// ---------------------------------------------------------------------------
// 2a) f32 -> f16 plain conversion (W)
// ---------------------------------------------------------------------------
__global__ void hfn_cvt_f16_kernel(const float* __restrict__ in,
                                   _Float16* __restrict__ out, int n)
{
    int i = blockIdx.x * blockDim.x + threadIdx.x;
    if (i < n) out[i] = (_Float16)in[i];
}

// ---------------------------------------------------------------------------
// 2b) h (N x C f32, row-major) -> hT (C x N f16), tiled via LDS so both the
//     global read and the global write are coalesced. 32x32 tiles.
// ---------------------------------------------------------------------------
__global__ __launch_bounds__(256)
void hfn_transpose_f16_kernel(const float* __restrict__ h,
                              _Float16* __restrict__ hT)
{
    __shared__ _Float16 tile[32][33];
    const int nBase = blockIdx.x * 32;   // NRES/32 blocks
    const int cBase = blockIdx.y * 32;   // CCH/32 blocks
    const int tx = threadIdx.x & 31;
    const int ty = threadIdx.x >> 5;     // 8 rows per pass
#pragma unroll
    for (int r = ty; r < 32; r += 8)
        tile[r][tx] = (_Float16)h[(size_t)(nBase + r) * CCH + cBase + tx];
    __syncthreads();
#pragma unroll
    for (int r = ty; r < 32; r += 8)
        hT[(size_t)(cBase + r) * NRES + nBase + tx] = tile[tx][r];
}

// ---------------------------------------------------------------------------
// 3) mask output: mask[n, m] = (sum |pos_n - nodepos_m|^3 < cutoff^3)
//    4 node columns per thread -> float4 stores (global_store_b128).
// ---------------------------------------------------------------------------
__global__ __launch_bounds__(256)
void hfn_mask_kernel(const float* __restrict__ pos,
                     const float* __restrict__ nodepos,
                     float4* __restrict__ maskout)
{
    size_t t = (size_t)blockIdx.x * 256 + threadIdx.x;  // t < N*M/4
    int n  = (int)(t >> 10);            // / (MNOD/4)
    int m0 = (int)(t & 1023) << 2;      // 4 consecutive m
    const float px = pos[3*n+0], py = pos[3*n+1], pz = pos[3*n+2];
    float4 r;
    float* rp = (float*)&r;
#pragma unroll
    for (int k = 0; k < 4; ++k) {
        int m = m0 + k;
        float dx = fabsf(px - nodepos[3*m+0]);
        float dy = fabsf(py - nodepos[3*m+1]);
        float dz = fabsf(pz - nodepos[3*m+2]);
        float d3 = dx*dx*dx + dy*dy*dy + dz*dz*dz;
        rp[k] = (d3 < CUT3) ? 1.0f : 0.0f;
    }
    maskout[t] = r;
}

// ---------------------------------------------------------------------------
// 4) agg[m,c] = sum_n mask[n,m] * h[n,c]  via v_wmma_f32_16x16x32_f16.
//    One wave owns a 16m x 128c strip (8 c-tiles): the register-computed
//    0/1 mask A-fragment is reused by 8 WMMAs per K-step.
//    A 16x32 f16 layout: K = j + (j&8) + hi*8, M = lane&15.
//    B 32x16 f16 as loaded here: K = j + hi*16, N = lane&15 (A built to match).
//    With hT[c][n], each B fragment is 32 contiguous bytes -> one v16h load
//    (2x global_load_b128). hT (4 MB) is L2-resident.
// ---------------------------------------------------------------------------
__global__ __launch_bounds__(32)
void hfn_agg_wmma_kernel(const float* __restrict__ pos,
                         const float* __restrict__ nodepos,
                         const _Float16* __restrict__ hT,
                         _Float16* __restrict__ aggf16)
{
    const int mTile = blockIdx.x;       // M/16 tiles
    const int cGrp  = blockIdx.y;       // C/128 groups (2)
    const int lane  = threadIdx.x;
    const int row   = lane & 15;
    const int hi    = lane >> 4;

    const int am = mTile * 16 + row;    // A-matrix row (node index)
    const float nx = nodepos[3*am+0];
    const float ny = nodepos[3*am+1];
    const float nz = nodepos[3*am+2];
    // per-lane B column base: channel c = cGrp*128 + ct*16 + row
    const _Float16* hcol0 = hT + (size_t)(cGrp * 128 + row) * NRES + hi * 16;

    v8f acc[8];
#pragma unroll
    for (int ct = 0; ct < 8; ++ct) acc[ct] = (v8f){};

    for (int kb = 0; kb < NRES; kb += 32) {
        // --- A fragment: 0/1 mask, exact in f16, computed in registers ---
        v16h a;
#pragma unroll
        for (int j = 0; j < 16; ++j) {
            const int K = j + hi * 16;            // contiguous K, matches B
            const int n = kb + K;
            float dx = fabsf(pos[3*n+0] - nx);
            float dy = fabsf(pos[3*n+1] - ny);
            float dz = fabsf(pos[3*n+2] - nz);
            float d3 = dx*dx*dx + dy*dy*dy + dz*dz*dz;
            a[j] = (_Float16)((d3 < CUT3) ? 1.0f : 0.0f);
        }
        // --- 8 B fragments: single 32-byte vector load each, shared A ---
#pragma unroll
        for (int ct = 0; ct < 8; ++ct) {
            v16h b = *(const v16h*)(hcol0 + (size_t)ct * 16 * NRES + kb);
            acc[ct] = __builtin_amdgcn_wmma_f32_16x16x32_f16(
                false, a, false, b, (short)0, acc[ct], false, false);
        }
    }
    // C/D layout: lanes 0-15 -> M=r, lanes 16-31 -> M=8+r; N = lane&15
#pragma unroll
    for (int ct = 0; ct < 8; ++ct) {
#pragma unroll
        for (int r = 0; r < 8; ++r) {
            int m = mTile * 16 + hi * 8 + r;
            aggf16[m * CCH + cGrp * 128 + ct * 16 + row] = (_Float16)acc[ct][r];
        }
    }
}

// ---------------------------------------------------------------------------
// 5) embed[m,o] = sum_c agg[m,c] * W[o,c] + b[o]   (nn.Linear, W.T matmul)
//    A fragment: two contiguous v8h halves; B fragment: one v16h load.
// ---------------------------------------------------------------------------
__global__ __launch_bounds__(32)
void hfn_linear_wmma_kernel(const _Float16* __restrict__ aggf16,
                            const _Float16* __restrict__ Wf16,
                            const float* __restrict__ bias,
                            float* __restrict__ out)
{
    const int mTile = blockIdx.x;
    const int oTile = blockIdx.y;
    const int lane  = threadIdx.x;
    const int row   = lane & 15;
    const int hi    = lane >> 4;

    const _Float16* arow = aggf16 + (size_t)(mTile * 16 + row) * CCH;
    const _Float16* wrow = Wf16 + (size_t)(oTile * 16 + row) * CCH;

    v8f acc = {};
    for (int kb = 0; kb < CCH; kb += 32) {
        // A: K = j + hi*16 -> 16 contiguous elements, matches B
        v8h alo = *(const v8h*)(arow + kb + hi * 16);       // K = hi*16 .. +7
        v8h ahi = *(const v8h*)(arow + kb + hi * 16 + 8);   // K = hi*16+8 ..
        v16h a = __builtin_shufflevector(alo, ahi,
                 0, 1, 2, 3, 4, 5, 6, 7, 8, 9, 10, 11, 12, 13, 14, 15);
        // B: K = j + hi*16 -> 16 contiguous elements of W row
        v16h b = *(const v16h*)(wrow + kb + hi * 16);       // B(K,N)=W[N][K]
        acc = __builtin_amdgcn_wmma_f32_16x16x32_f16(
            false, a, false, b, (short)0, acc, false, false);
    }
    const float bv = bias[oTile * 16 + row];
#pragma unroll
    for (int r = 0; r < 8; ++r) {
        int m = mTile * 16 + hi * 8 + r;
        out[m * CCH + oTile * 16 + row] = acc[r] + bv;
    }
}

// ---------------------------------------------------------------------------
extern "C" void kernel_launch(void* const* d_in, const int* in_sizes, int n_in,
                              void* d_out, int out_size, void* d_ws, size_t ws_size,
                              hipStream_t stream)
{
    const float* h   = (const float*)d_in[0]; // (1, 8192, 256)
    const float* pos = (const float*)d_in[1]; // (8192, 3)
    const float* W   = (const float*)d_in[2]; // (256, 256)
    const float* b   = (const float*)d_in[3]; // (256,)

    float* out = (float*)d_out;
    float* out_embed = out;                          // M*C
    float* out_pos   = out + (size_t)MNOD * CCH;     // M*3
    float* out_mask  = out_pos + (size_t)MNOD * 3;   // N*M

    // workspace layout (bytes)
    char* ws = (char*)d_ws;
    float*    nodepos = (float*)   (ws + 0);                 // M*3*4   = 49152
    _Float16* hT      = (_Float16*)(ws + 49152);             // C*N*2   = 4194304
    _Float16* Wf16    = (_Float16*)(ws + 49152 + 4194304);   // C*C*2   = 131072
    _Float16* aggf16  = (_Float16*)(ws + 49152 + 4194304 + 131072); // M*C*2

    // conversions (independent of FPS)
    dim3 gT(NRES / 32, CCH / 32);
    hfn_transpose_f16_kernel<<<gT, 256, 0, stream>>>(h, hT);
    hfn_cvt_f16_kernel<<<(CCH * CCH + 255) / 256, 256, 0, stream>>>(W, Wf16, CCH * CCH);

    // farthest-point sampling -> node positions (ascending index order)
    hfn_fps_kernel<<<1, 1024, 0, stream>>>(pos, nodepos, out_pos);

    // mask output (f32 0/1), float4 stores
    hfn_mask_kernel<<<(unsigned)(((size_t)NRES * MNOD) / (4 * 256)), 256, 0, stream>>>(
        pos, nodepos, (float4*)out_mask);

    // agg = mask^T @ h   (WMMA f16 -> f32), 16m x 128c strip per wave
    dim3 gAgg(MNOD / 16, CCH / 128);
    hfn_agg_wmma_kernel<<<gAgg, 32, 0, stream>>>(pos, nodepos, hT, aggf16);

    // embed = agg @ W^T + b   (WMMA f16 -> f32)
    dim3 gLin(MNOD / 16, CCH / 16);
    hfn_linear_wmma_kernel<<<gLin, 32, 0, stream>>>(aggf16, Wf16, b, out_embed);
}
